// VsmNormer_82660940579498
// MI455X (gfx1250) — matrix operand order repaired and checked by reading, not compiled, not run on hardware
//
#include <hip/hip_runtime.h>
#include <hip/hip_bf16.h>

// ---- problem constants (from reference) ----
#define VOCAB 50000
#define D 200
#define NCONCEPTS 100000
#define BATCH 256
#define LTOK 8
#define KP 224          // D padded to 7 * 32 (bf16 WMMA K=32 steps)
#define KSTEPS 7
#define NTILE 64        // concepts per workgroup

typedef __attribute__((ext_vector_type(16))) __bf16 v16bf;
typedef __attribute__((ext_vector_type(8)))  float  v8f;
typedef __attribute__((ext_vector_type(4)))  float  f32x4;

union ABfrag { v16bf v; f32x4 f[2]; };

// monotone mapping float -> uint so unsigned max == float max
__device__ __forceinline__ unsigned int encf(float f) {
    unsigned int u = __float_as_uint(f);
    return (u & 0x80000000u) ? ~u : (u | 0x80000000u);
}

// ---------------------------------------------------------------------------
// Kernel 1: mention_rep = mean(word_embedding[ids], axis=1); row-normalize;
// store bf16 A [256 x 224] (zero padded). Also init global packed max slots.
// One block per mention row.
// ---------------------------------------------------------------------------
__global__ __launch_bounds__(256) void build_mention(
    const int* __restrict__ ids, const float* __restrict__ we,
    __bf16* __restrict__ Abf, unsigned long long* __restrict__ gpacked)
{
    const int m = blockIdx.x;
    const int t = threadIdx.x;
    if (t == 0) gpacked[m] = 0ULL;

    int id[LTOK];
#pragma unroll
    for (int l = 0; l < LTOK; ++l) id[l] = ids[m * LTOK + l];

    float s = 0.f;
    if (t < D) {
#pragma unroll
        for (int l = 0; l < LTOK; ++l) s += we[id[l] * D + t];
        s *= (1.0f / LTOK);
    }

    __shared__ float red[256];
    red[t] = (t < D) ? s * s : 0.f;
    __syncthreads();
    for (int off = 128; off > 0; off >>= 1) {
        if (t < off) red[t] += red[t + off];
        __syncthreads();
    }
    const float rinv = 1.0f / fmaxf(sqrtf(red[0]), 1e-8f);
    if (t < KP) Abf[m * KP + t] = (__bf16)(s * rinv);   // t>=D contributes 0
}

// ---------------------------------------------------------------------------
// Kernel 2: per block: stage 64 concepts as bf16 into LDS, compute 1/||c||,
// 8 waves x (2 M-tiles x 4 N-tiles) of v_wmma_f32_16x16x32_bf16 over K=224.
// Per lane: keep 8 packed (value<<32 | ~index) running maxima in registers
// across the nt loop (branchless validity masking), then 8 ds_max_u64 LDS
// atomics per M-tile; one global atomicMax(u64) per output row per block.
// ---------------------------------------------------------------------------
__global__ __launch_bounds__(256) void sim_kernel(
    const float* __restrict__ dict, const __bf16* __restrict__ Abf,
    unsigned long long* __restrict__ gpacked, int Ntot)
{
    __shared__ __attribute__((aligned(16))) __bf16 sdict[NTILE * KP];
    __shared__ float snrm[NTILE];
    __shared__ float scinv[NTILE];
    __shared__ unsigned long long sbest[BATCH];

    const int tid = threadIdx.x;
    const int n0  = blockIdx.x * NTILE;

    sbest[tid] = 0ULL;
    if (tid < NTILE) snrm[tid] = 0.f;
    __syncthreads();

    // ---- stage fp32 -> bf16 tile into LDS (coalesced 224-lane rows) ----
    if (tid < KP) {
#pragma unroll 1
        for (int r = 0; r < NTILE; ++r) {
            const int c = n0 + r;
            float v = 0.f;
            if (tid < D && c < Ntot) v = dict[(long)c * D + tid];
            sdict[r * KP + tid] = (__bf16)v;
        }
    }
    __syncthreads();

    // ---- concept norms: 4 threads/row, 7 x ds_load_b128 (56 bf16) each ----
    {
        const int r = tid >> 2, q = tid & 3;
        float s = 0.f;
        const f32x4* pr = reinterpret_cast<const f32x4*>(&sdict[r * KP + q * 56]);
#pragma unroll
        for (int i = 0; i < 7; ++i) {
            union { f32x4 f; __bf16 h[8]; } u;
            u.f = pr[i];
#pragma unroll
            for (int j = 0; j < 8; ++j) {
                float v = (float)u.h[j];
                s += v * v;       // padded dims are exactly zero
            }
        }
        atomicAdd(&snrm[r], s);
    }
    __syncthreads();
    if (tid < NTILE) {
        const int c = n0 + tid;
        scinv[tid] = (c < Ntot) ? (1.0f / fmaxf(sqrtf(snrm[tid]), 1e-8f)) : 0.f;
    }
    __syncthreads();

    // ---- WMMA phase ----
    const int wv   = tid >> 5;
    const int lane = tid & 31;
    const int r16  = lane & 15;   // A: row-in-tile / B: col-in-tile / D: N index
    const int h    = lane >> 4;   // half-wave select

#pragma unroll 1
    for (int mtv = 0; mtv < 2; ++mtv) {
        const int m0 = (wv * 2 + mtv) * 16;

        // A fragments for all 7 K-steps (two b128 global loads each)
        ABfrag af[KSTEPS];
#pragma unroll
        for (int ks = 0; ks < KSTEPS; ++ks) {
            const f32x4* pa = reinterpret_cast<const f32x4*>(
                Abf + (m0 + r16) * KP + ks * 32 + h * 16);
            af[ks].f[0] = pa[0];
            af[ks].f[1] = pa[1];
        }

        // running best per output row held in this lane (VGPR v -> M = m0+v+8h)
        unsigned long long bestv[8];
#pragma unroll
        for (int v = 0; v < 8; ++v) bestv[v] = 0ULL;

#pragma unroll 1
        for (int nt = 0; nt < 4; ++nt) {
            v8f acc = {};
#pragma unroll
            for (int ks = 0; ks < KSTEPS; ++ks) {
                ABfrag bfr;
                const f32x4* pb = reinterpret_cast<const f32x4*>(
                    &sdict[(nt * 16 + r16) * KP + ks * 32 + h * 16]);
                bfr.f[0] = pb[0];
                bfr.f[1] = pb[1];
                acc = __builtin_amdgcn_wmma_f32_16x16x32_bf16(
                    false, af[ks].v, false, bfr.v, (short)0, acc, false, false);
            }

            const int nloc   = nt * 16 + r16;
            const int gc     = n0 + nloc;
            const float cinv = scinv[nloc];
            // branchless validity: vm = all-ones iff gc < Ntot
            const unsigned int vm =
                (unsigned int)((int)(gc - Ntot) >> 31);
            const unsigned int idxbits = (~(unsigned int)gc) & vm;

#pragma unroll
            for (int v = 0; v < 8; ++v) {
                const float s = acc[v] * cinv;
                const unsigned long long pk =
                    (((unsigned long long)(encf(s) & vm)) << 32) | idxbits;
                bestv[v] = (pk > bestv[v]) ? pk : bestv[v];
            }
        }

        // 8 no-return ds_max_u64 atomics per lane (16-way row sharing)
#pragma unroll
        for (int v = 0; v < 8; ++v) {
            atomicMax(&sbest[m0 + v + 8 * h], bestv[v]);
        }
    }
    __syncthreads();
    atomicMax(&gpacked[tid], sbest[tid]);
}

// ---------------------------------------------------------------------------
// Kernel 3: decode packed (value, ~index) -> d_out
// ---------------------------------------------------------------------------
__global__ void finalize(const unsigned long long* __restrict__ gpacked,
                         float* __restrict__ out)
{
    const int t = threadIdx.x;
    const unsigned long long p = gpacked[t];
    const unsigned int hi = (unsigned int)(p >> 32);
    const unsigned int lo = (unsigned int)p;
    const unsigned int ub = (hi & 0x80000000u) ? (hi & 0x7fffffffu) : ~hi;
    out[t] = __uint_as_float(ub);
    ((int*)out)[BATCH + t] = (int)(~lo);
}

extern "C" void kernel_launch(void* const* d_in, const int* in_sizes, int n_in,
                              void* d_out, int out_size, void* d_ws, size_t ws_size,
                              hipStream_t stream) {
    const int*   mids = (const int*)d_in[0];     // [256, 8]
    const float* we   = (const float*)d_in[1];   // [50000, 200]
    const float* dict = (const float*)d_in[2];   // [100000, 200]

    __bf16* Abf = (__bf16*)d_ws;                                   // 256*224*2 B
    unsigned long long* gpacked =
        (unsigned long long*)((char*)d_ws + (size_t)BATCH * KP * 2); // 256*8 B

    build_mention<<<BATCH, 256, 0, stream>>>(mids, we, Abf, gpacked);

    const int nblk = (NCONCEPTS + NTILE - 1) / NTILE;  // 1563
    sim_kernel<<<nblk, 256, 0, stream>>>(dict, Abf, gpacked, NCONCEPTS);

    finalize<<<1, BATCH, 0, stream>>>(gpacked, (float*)d_out);
}
